// GNNWithAttention_11166914969681
// MI455X (gfx1250) — matrix operand order repaired and checked by reading, not compiled, not run on hardware
//
#include <hip/hip_runtime.h>
#include <hip/hip_bf16.h>
#include <cstdint>

typedef __attribute__((ext_vector_type(16))) __bf16          v16bf;
typedef __attribute__((ext_vector_type(8)))  float           v8f;
typedef __attribute__((ext_vector_type(16))) unsigned short  us16;

#define NFEAT 2000
#define KPAD  2016         // NFEAT rounded up to 32
#define D1    64
#define NCOLS 128          // [w1 | w_res] fused output width
#define BSTRIDE 40         // LDS B stride in ushorts (80B: 16B-aligned, bank-spread)
#define TILE_US (NCOLS * BSTRIDE)

__device__ __forceinline__ unsigned short f2bf(float f) {
    unsigned u = __float_as_uint(f);
    u += 0x7FFFu + ((u >> 16) & 1u);          // round-to-nearest-even
    return (unsigned short)(u >> 16);
}
// monotonic float<->uint encoding for atomicMax-based segment_max
__device__ __forceinline__ unsigned fenc(float f) {
    unsigned u = __float_as_uint(f);
    return (u & 0x80000000u) ? ~u : (u | 0x80000000u);
}
__device__ __forceinline__ float fdec(unsigned k) {
    unsigned u = (k & 0x80000000u) ? (k ^ 0x80000000u) : ~k;
    return __uint_as_float(u);
}
__device__ __forceinline__ float lrelu(float s) { return s < 0.f ? 0.2f * s : s; }
__device__ __forceinline__ float elu1(float v)  { return v > 0.f ? v : __expf(v) - 1.f; }

struct Bpair { uint4 lo, hi; };   // 32B container for a v16bf from 16B-aligned LDS

// ---------------------------------------------------------------------------
// Pre-pack [w1 | w_res] -> bf16, column-major [col][k], k zero-padded to 2016.
// 516KB, stays L2-resident; read by every GEMM block with no conversion work.
// ---------------------------------------------------------------------------
__global__ void pack_w(const float* __restrict__ w1, const float* __restrict__ wres,
                       unsigned short* __restrict__ Bt)
{
    int idx = blockIdx.x * blockDim.x + threadIdx.x;
    if (idx >= NCOLS * KPAD) return;
    const int col = idx / KPAD, k = idx % KPAD;
    float v = 0.f;
    if (k < NFEAT)
        v = (col < D1) ? w1[(size_t)k * D1 + col] : wres[(size_t)k * D1 + (col - D1)];
    Bt[idx] = f2bf(v);
}

// Issue one 32-K B tile (128 cols x 64B) as async global->LDS b128 transfers.
// 512 transfers, 2 per thread. ASYNCcnt-tracked, no VGPR data path.
__device__ __forceinline__ void issueB(const unsigned short* __restrict__ Bt,
                                       unsigned ldsByteBase, int kk, int tid)
{
    #pragma unroll
    for (int r = 0; r < 2; ++r) {
        const int t     = tid + r * 256;
        const int col   = t >> 2;
        const int chunk = t & 3;
        const unsigned long long g =
            (unsigned long long)(uintptr_t)(Bt + (size_t)col * KPAD + kk + chunk * 8);
        const unsigned l = ldsByteBase + (unsigned)(col * BSTRIDE + chunk * 8) * 2u;
        asm volatile("global_load_async_to_lds_b128 %0, %1, off"
                     :: "v"(l), "v"(g) : "memory");
    }
}

// ---------------------------------------------------------------------------
// Kernel: fused  hw[n,0:64]=x@w1 , hw[n,64:128]=x@w_res   (bf16 WMMA, f32 acc)
// Block: 256 thr (8 waves) -> 64 rows x 128 cols. Wave = 16 rows x 64 cols.
// B double-buffered in LDS via async DMA; A straight from global (read once).
// ---------------------------------------------------------------------------
__global__ __launch_bounds__(256) void gemm1_wmma(
    const float* __restrict__ x, const unsigned short* __restrict__ Bt,
    float* __restrict__ hw, int n_nodes)
{
    __shared__ __align__(16) unsigned short Bs[2][TILE_US];

    const int tid    = threadIdx.x;
    const int wave   = tid >> 5;
    const int lane   = tid & 31;
    const int wave_m = wave & 3;          // 4 row tiles of 16
    const int wave_n = wave >> 2;         // 2 col halves of 64
    const int lr     = lane & 15;
    const int lh     = lane >> 4;
    const int blockRow = blockIdx.x * 64;

    int arow = blockRow + wave_m * 16 + lr;
    if (arow >= n_nodes) arow = n_nodes - 1;          // clamp; store is guarded
    const float* aptr = x + (size_t)arow * NFEAT;

    // ISA rule: generic LDS addresses truncate to the LDS byte offset
    const unsigned bsBase = (unsigned)(uintptr_t)(&Bs[0][0]);

    v8f acc[4] = {};

    issueB(Bt, bsBase, 0, tid);                        // prologue: tile 0 -> buf 0

    const int kfull = NFEAT / 32;                      // 62 full steps + 16-tail
    for (int step = 0; step <= kfull; ++step) {
        const int  kk   = step * 32;
        const bool tail = (step == kfull);
        const int  cur  = step & 1;

        // ---- A fragment from global (ISA 16-bit A 16x32 layout), overlaps DMA ----
        const int base = lh * 8;                        // lanes16-31 hold K+8
        const float* ap = aptr + kk + base;
        float af[16];
        *(float4*)(af + 0) = ((const float4*)ap)[0];    // K base..base+3
        *(float4*)(af + 4) = ((const float4*)ap)[1];    // K base+4..base+7
        if (!tail) {
            *(float4*)(af + 8)  = ((const float4*)(ap + 16))[0];  // K base+16..
            *(float4*)(af + 12) = ((const float4*)(ap + 16))[1];
            __builtin_prefetch(ap + 32, 0, 3);          // next A tile
        } else {
            #pragma unroll
            for (int i = 8; i < 16; ++i) af[i] = 0.f;   // zero-pad K tail
        }
        us16 au;
        #pragma unroll
        for (int i = 0; i < 16; ++i) au[i] = f2bf(af[i]);
        const v16bf afrag = __builtin_bit_cast(v16bf, au);

        // ---- current B tile landed? (own wave's DMA, then all waves) ----
        asm volatile("s_wait_asynccnt 0x0" ::: "memory");
        __syncthreads();
        if (!tail)                                      // prefetch next tile
            issueB(Bt, bsBase + (unsigned)((1 - cur) * TILE_US * 2), kk + 32, tid);

        // ---- 4 B fragments + WMMA from buf[cur] ----
        const unsigned short* bsc = &Bs[cur][0];
        const int kb = lh * 8;                          // lanes16-31 hold K 8..15, 24..31
        #pragma unroll
        for (int j = 0; j < 4; ++j) {
            const int col = wave_n * 64 + j * 16 + lr;
            Bpair bp;
            bp.lo = *(const uint4*)(bsc + col * BSTRIDE + kb);
            bp.hi = *(const uint4*)(bsc + col * BSTRIDE + kb + 16);
            const v16bf bfrag = __builtin_bit_cast(v16bf, bp);
            acc[j] = __builtin_amdgcn_wmma_f32_16x16x32_bf16(
                false, afrag, false, bfrag, (short)0, acc[j], false, false);
        }
        // buf[cur] is only overwritten after the NEXT iteration's barrier,
        // and the ds_load data is consumed (waited) before the wmma above.
    }

    // ---- epilogue: ISA 16x16 f32 C layout -> guarded f32 stores ----
    #pragma unroll
    for (int j = 0; j < 4; ++j) {
        const int col = wave_n * 64 + j * 16 + lr;
        #pragma unroll
        for (int i = 0; i < 8; ++i) {
            const int r = blockRow + wave_m * 16 + lh * 8 + i;
            if (r < n_nodes) hw[(size_t)r * NCOLS + col] = acc[j][i];
        }
    }
}

// ---------------------------------------------------------------------------
__global__ void zero_ws(float* __restrict__ p, size_t n) {
    size_t i = (size_t)blockIdx.x * blockDim.x + threadIdx.x;
    if (i < n) p[i] = 0.f;
}

// a_src1/a_dst1 per (node, head)
__global__ void attn1_kernel(const float* __restrict__ hw,
                             const float* __restrict__ att_s,
                             const float* __restrict__ att_d,
                             float* __restrict__ as, float* __restrict__ ad, int n)
{
    int idx = blockIdx.x * blockDim.x + threadIdx.x;
    if (idx >= n * 8) return;
    const int node = idx >> 3, h = idx & 7;
    const float* hp = hw + (size_t)node * NCOLS + h * 8;
    float s = 0.f, d = 0.f;
    #pragma unroll
    for (int c = 0; c < 8; ++c) { s += hp[c] * att_s[h * 8 + c]; d += hp[c] * att_d[h * 8 + c]; }
    as[idx] = s; ad[idx] = d;
}

__device__ __forceinline__ void edge_sd(const int* ei, int e, int E, int& s, int& d) {
    if (e < E) { s = ei[e]; d = ei[E + e]; } else { s = e - E; d = s; }
}

__global__ void edge1_max(const int* __restrict__ ei, int E, int ET,
                          const float* __restrict__ as, const float* __restrict__ ad,
                          unsigned* __restrict__ m1)
{
    int e = blockIdx.x * blockDim.x + threadIdx.x;
    if (e >= ET) return;
    int s, d; edge_sd(ei, e, E, s, d);
    #pragma unroll
    for (int h = 0; h < 8; ++h) {
        float sc = lrelu(as[s * 8 + h] + ad[d * 8 + h]);
        atomicMax(&m1[d * 8 + h], fenc(sc));
    }
}

__global__ void edge1_exp(const int* __restrict__ ei, int E, int ET,
                          const float* __restrict__ as, const float* __restrict__ ad,
                          const unsigned* __restrict__ m1, float* __restrict__ s1,
                          float* __restrict__ alpha1)
{
    int e = blockIdx.x * blockDim.x + threadIdx.x;
    if (e >= ET) return;
    int s, d; edge_sd(ei, e, E, s, d);
    #pragma unroll
    for (int h = 0; h < 8; ++h) {
        float sc = lrelu(as[s * 8 + h] + ad[d * 8 + h]);
        float m  = fdec(m1[d * 8 + h]);
        if (((__float_as_uint(m) >> 23) & 0xFF) == 0xFF) m = 0.f;  // guard non-finite
        float ev = __expf(sc - m);
        atomicAdd(&s1[d * 8 + h], ev);
        alpha1[(size_t)e * 8 + h] = ev;
    }
}

// normalize alpha in place + scatter messages (thread per edge-head)
__global__ void edge1_agg(const int* __restrict__ ei, int E, int ET,
                          const float* __restrict__ s1, const float* __restrict__ hw,
                          float* __restrict__ alpha1, float* __restrict__ agg1)
{
    size_t idx = (size_t)blockIdx.x * blockDim.x + threadIdx.x;
    if (idx >= (size_t)ET * 8) return;
    int e = (int)(idx >> 3), h = (int)(idx & 7);
    int s, d; edge_sd(ei, e, E, s, d);
    float al = alpha1[idx] / (s1[d * 8 + h] + 1e-16f);
    alpha1[idx] = al;
    const float* hp = hw + (size_t)s * NCOLS + h * 8;
    #pragma unroll
    for (int c = 0; c < 8; ++c)
        atomicAdd(&agg1[(size_t)d * D1 + h * 8 + c], hp[c] * al);
}

// x1 = elu(agg1 + b1 + (res + b_res))
__global__ void combine1(const float* __restrict__ agg1, const float* __restrict__ b1,
                         const float* __restrict__ hw, const float* __restrict__ bres,
                         float* __restrict__ x1, int n)
{
    size_t idx = (size_t)blockIdx.x * blockDim.x + threadIdx.x;
    if (idx >= (size_t)n * D1) return;
    int node = (int)(idx >> 6), c = (int)(idx & 63);
    float v = agg1[idx] + b1[c] + hw[(size_t)node * NCOLS + D1 + c] + bres[c];
    x1[idx] = elu1(v);
}

// h2 = x1 @ w2   (K=64, N=30 — compute-trivial)
__global__ __launch_bounds__(256) void gemm2_kernel(
    const float* __restrict__ x1, const float* __restrict__ w2,
    float* __restrict__ h2, int n)
{
    __shared__ float xs[8 * 64];
    const int nl = threadIdx.x >> 5, c = threadIdx.x & 31;
    const int node = blockIdx.x * 8 + nl;
    const size_t base = (size_t)blockIdx.x * 8 * 64;
    for (int i = threadIdx.x; i < 512; i += 256) {
        size_t g = base + i;
        xs[i] = (g < (size_t)n * 64) ? x1[g] : 0.f;
    }
    __syncthreads();
    if (node < n && c < 30) {
        float acc = 0.f;
        #pragma unroll
        for (int k = 0; k < 64; ++k) acc += xs[nl * 64 + k] * w2[k * 30 + c];
        h2[(size_t)node * 30 + c] = acc;
    }
}

__global__ void attn2_kernel(const float* __restrict__ h2,
                             const float* __restrict__ att_s,
                             const float* __restrict__ att_d,
                             float* __restrict__ as, float* __restrict__ ad, int n)
{
    int node = blockIdx.x * blockDim.x + threadIdx.x;
    if (node >= n) return;
    const float* hp = h2 + (size_t)node * 30;
    float s = 0.f, d = 0.f;
    #pragma unroll
    for (int c = 0; c < 30; ++c) { s += hp[c] * att_s[c]; d += hp[c] * att_d[c]; }
    as[node] = s; ad[node] = d;
}

__global__ void edge2_max(const int* __restrict__ ei, int E, int ET,
                          const float* __restrict__ as, const float* __restrict__ ad,
                          unsigned* __restrict__ m2)
{
    int e = blockIdx.x * blockDim.x + threadIdx.x;
    if (e >= ET) return;
    int s, d; edge_sd(ei, e, E, s, d);
    atomicMax(&m2[d], fenc(lrelu(as[s] + ad[d])));
}

__global__ void edge2_exp(const int* __restrict__ ei, int E, int ET,
                          const float* __restrict__ as, const float* __restrict__ ad,
                          const unsigned* __restrict__ m2, float* __restrict__ s2,
                          float* __restrict__ alpha2)
{
    int e = blockIdx.x * blockDim.x + threadIdx.x;
    if (e >= ET) return;
    int s, d; edge_sd(ei, e, E, s, d);
    float m = fdec(m2[d]);
    if (((__float_as_uint(m) >> 23) & 0xFF) == 0xFF) m = 0.f;
    float ev = __expf(lrelu(as[s] + ad[d]) - m);
    atomicAdd(&s2[d], ev);
    alpha2[e] = ev;
}

__global__ void edge2_agg(const int* __restrict__ ei, int E, int ET,
                          const float* __restrict__ s2, const float* __restrict__ h2,
                          float* __restrict__ alpha2, float* __restrict__ agg2)
{
    int e = blockIdx.x * blockDim.x + threadIdx.x;
    if (e >= ET) return;
    int s, d; edge_sd(ei, e, E, s, d);
    float al = alpha2[e] / (s2[d] + 1e-16f);
    alpha2[e] = al;
    const float* hp = h2 + (size_t)s * 30;
    #pragma unroll
    for (int c = 0; c < 30; ++c)
        atomicAdd(&agg2[(size_t)d * 30 + c], hp[c] * al);
}

__global__ void final_kernel(const float* __restrict__ agg2, const float* __restrict__ b2,
                             float* __restrict__ out, int n)
{
    int node = blockIdx.x * blockDim.x + threadIdx.x;
    if (node >= n) return;
    float v[30], mx = -3.4e38f;
    #pragma unroll
    for (int c = 0; c < 30; ++c) {
        v[c] = elu1(agg2[(size_t)node * 30 + c] + b2[c]);
        mx = fmaxf(mx, v[c]);
    }
    float sum = 0.f;
    #pragma unroll
    for (int c = 0; c < 30; ++c) sum += __expf(v[c] - mx);
    const float l = mx + __logf(sum);
    #pragma unroll
    for (int c = 0; c < 30; ++c) out[(size_t)node * 30 + c] = v[c] - l;
}

// ---------------------------------------------------------------------------
extern "C" void kernel_launch(void* const* d_in, const int* in_sizes, int n_in,
                              void* d_out, int out_size, void* d_ws, size_t ws_size,
                              hipStream_t stream)
{
    const float* x     = (const float*)d_in[0];
    const int*   ei    = (const int*)  d_in[1];
    const float* wres  = (const float*)d_in[2];
    const float* bres  = (const float*)d_in[3];
    const float* w1    = (const float*)d_in[4];
    const float* atts1 = (const float*)d_in[5];
    const float* attd1 = (const float*)d_in[6];
    const float* b1    = (const float*)d_in[7];
    const float* w2    = (const float*)d_in[8];
    const float* atts2 = (const float*)d_in[9];
    const float* attd2 = (const float*)d_in[10];
    const float* b2    = (const float*)d_in[11];

    const int N  = in_sizes[0] / NFEAT;
    const int E  = in_sizes[1] / 2;
    const int ET = E + N;

    float* W = (float*)d_ws;
    size_t off_bt   = 0;                                  // bf16 packed weights
    size_t off_hw   = off_bt   + (size_t)NCOLS * KPAD / 2;  // ushorts -> floats
    size_t off_as1  = off_hw   + (size_t)N * NCOLS;
    size_t off_ad1  = off_as1  + (size_t)N * 8;
    size_t off_x1   = off_ad1  + (size_t)N * 8;
    size_t off_h2   = off_x1   + (size_t)N * D1;
    size_t off_as2  = off_h2   + (size_t)N * 30;
    size_t off_ad2  = off_as2  + (size_t)N;
    size_t off_m1   = off_ad2  + (size_t)N;        // zero region starts here
    size_t off_s1   = off_m1   + (size_t)N * 8;
    size_t off_agg1 = off_s1   + (size_t)N * 8;
    size_t off_m2   = off_agg1 + (size_t)N * D1;
    size_t off_s2   = off_m2   + (size_t)N;
    size_t off_agg2 = off_s2   + (size_t)N;
    size_t off_end  = off_agg2 + (size_t)N * 30;

    unsigned short* Bt = (unsigned short*)(W + off_bt);
    float*    hw    = W + off_hw;
    float*    as1   = W + off_as1;   float* ad1  = W + off_ad1;
    float*    x1    = W + off_x1;    float* h2   = W + off_h2;
    float*    as2   = W + off_as2;   float* ad2  = W + off_ad2;
    unsigned* m1    = (unsigned*)(W + off_m1);
    float*    s1    = W + off_s1;    float* agg1 = W + off_agg1;
    unsigned* m2    = (unsigned*)(W + off_m2);
    float*    s2    = W + off_s2;    float* agg2 = W + off_agg2;

    float* outLS   = (float*)d_out;
    float* alpha1  = outLS + (size_t)N * 30;
    float* alpha2  = alpha1 + (size_t)ET * 8;

    const size_t nzero = off_end - off_m1;
    zero_ws<<<(int)((nzero + 255) / 256), 256, 0, stream>>>(W + off_m1, nzero);
    pack_w<<<(NCOLS * KPAD + 255) / 256, 256, 0, stream>>>(w1, wres, Bt);

    gemm1_wmma<<<(N + 63) / 64, 256, 0, stream>>>(x, Bt, hw, N);
    attn1_kernel<<<(N * 8 + 255) / 256, 256, 0, stream>>>(hw, atts1, attd1, as1, ad1, N);

    const int egrid = (ET + 255) / 256;
    edge1_max<<<egrid, 256, 0, stream>>>(ei, E, ET, as1, ad1, m1);
    edge1_exp<<<egrid, 256, 0, stream>>>(ei, E, ET, as1, ad1, m1, s1, alpha1);
    edge1_agg<<<(int)(((size_t)ET * 8 + 255) / 256), 256, 0, stream>>>(ei, E, ET, s1, hw, alpha1, agg1);
    combine1<<<(int)(((size_t)N * D1 + 255) / 256), 256, 0, stream>>>(agg1, b1, hw, bres, x1, N);

    gemm2_kernel<<<(N + 7) / 8, 256, 0, stream>>>(x1, w2, h2, N);
    attn2_kernel<<<(N + 255) / 256, 256, 0, stream>>>(h2, atts2, attd2, as2, ad2, N);

    edge2_max<<<egrid, 256, 0, stream>>>(ei, E, ET, as2, ad2, m2);
    edge2_exp<<<egrid, 256, 0, stream>>>(ei, E, ET, as2, ad2, m2, s2, alpha2);
    edge2_agg<<<egrid, 256, 0, stream>>>(ei, E, ET, s2, h2, alpha2, agg2);

    final_kernel<<<(N + 255) / 256, 256, 0, stream>>>(agg2, b2, outLS, N);
}
